// CustomLSTM_16913581212002
// MI455X (gfx1250) — compile-verified, matrix operand order checked by reading, not run on hardware
//
#include <hip/hip_runtime.h>
#include <hip/hip_bf16.h>

#define B_ 32
#define S_ 512
#define I_ 1024
#define H_ 512
#define G3 1536           // 3*H
#define MROWS (B_ * S_)   // 16384

// LDS tile row stride in dwords (64 bf16 data + 4 dwords pad -> conflict-free, 16B aligned)
#define LROW 36

typedef __attribute__((ext_vector_type(16))) __bf16 v16bf;
typedef __attribute__((ext_vector_type(8)))  float  v8f;
typedef unsigned int  u32x4 __attribute__((ext_vector_type(4)));
typedef int           i32x8 __attribute__((ext_vector_type(8)));
typedef int           i32x4 __attribute__((ext_vector_type(4)));
typedef unsigned long long u64;

__device__ __forceinline__ unsigned short f32_to_bf16(float f) {
    unsigned u = __float_as_uint(f);
    unsigned rnd = 0x7FFFu + ((u >> 16) & 1u);
    return (unsigned short)((u + rnd) >> 16);
}

__device__ __forceinline__ float sigmoidf_(float x) {
    return 1.0f / (1.0f + __expf(-x));
}
__device__ __forceinline__ float tanhf_(float x) {
    return 1.0f - 2.0f / (__expf(2.0f * x) + 1.0f);   // saturates correctly
}

// 16x32 bf16 WMMA fragment (A-style; B mirrored with N across lanes).
// rowbase: dword ptr to this lane's K-contiguous row; kd0 = k0/2.
// ISA layout: lane group g = lane[4]; VGPR v<4: K = 8g+2v,+1 ; v>=4: K = 16+8g+2(v-4),+1
// -> per lane: dwords [kd0+4g .. +3] and [kd0+4g+8 .. +3]: two aligned 16B runs.
__device__ __forceinline__ v16bf load_frag(const unsigned* rowbase, int kd0) {
    int g  = (threadIdx.x >> 4) & 1;
    int kd = kd0 + g * 4;
    union { v16bf v; uint4 q[2]; } f;
    f.q[0] = *(const uint4*)(rowbase + kd);
    f.q[1] = *(const uint4*)(rowbase + kd + 8);
    return f.v;
}

__device__ __forceinline__ v8f wmma_bf16(v16bf a, v16bf b, v8f c) {
    return __builtin_amdgcn_wmma_f32_16x16x32_bf16(false, a, false, b, (short)0, c, false, false);
}

// ------------------------------------------------------------------ TDM issue
// 2D tile load global->LDS via Tensor Data Mover (D# per CDNA5 ISA).
// data_size = 2 bytes (bf16). LDS pad: +4 dwords every 32 dwords (one 64-elem row)
// -> LDS row stride 36 dwords (144B), bank-conflict-free and 16B aligned.
__device__ __forceinline__ void tdm_load_2d_bf16(unsigned lds_addr, u64 gaddr,
                                                 unsigned tile_d0, unsigned tile_d1,
                                                 unsigned tensor_d0, unsigned tensor_d1,
                                                 u64 row_stride_elems) {
    u32x4 g0;
    g0[0] = 1u;                                   // count=1, user mode, no gather
    g0[1] = lds_addr;                             // LDS byte address
    g0[2] = (unsigned)gaddr;                      // global_addr[31:0]
    g0[3] = (unsigned)((gaddr >> 32) & 0x01FFFFFFull) | (2u << 30);  // [56:32] | type=2
    union { i32x8 v; u64 q[4]; } g1;
    // [15:0] wg_mask=0 | [17:16] data_size=1(2B) | [20] pad_en=1
    // | [24:22] pad_interval=4 (32 dwords) | [31:25] pad_amount=3 (4 dwords)
    // | [47:32] bar_addr=0 | [79:48] tensor_dim0 | [111:80] tensor_dim1
    // | [127:112] tile_dim0 | [143:128] tile_dim1 | [159:144] tile_dim2=0
    // | [207:160] dim0_stride | [255:208] dim1_stride=0
    g1.q[0] = (1ull << 16) | (1ull << 20) | (4ull << 22) | (3ull << 25)
            | ((u64)(tensor_d0 & 0xFFFFu) << 48);
    g1.q[1] = ((u64)tensor_d0 >> 16) | ((u64)tensor_d1 << 16) | ((u64)tile_d0 << 48);
    g1.q[2] = (u64)tile_d1 | (row_stride_elems << 32);
    g1.q[3] = (row_stride_elems >> 32) & 0xFFFFull;
    i32x4 z4 = {};
    i32x8 z8 = {};
    __builtin_amdgcn_tensor_load_to_lds(g0, g1.v, z4, z4, z8, 0);
}

// ---------------------------------------------------------------- conversions
__global__ void cvt_f32_bf16(const float* __restrict__ in, unsigned short* __restrict__ out, int n) {
    for (int i = blockIdx.x * blockDim.x + threadIdx.x; i < n; i += gridDim.x * blockDim.x)
        out[i] = f32_to_bf16(in[i]);
}

__global__ void init_state(const float* __restrict__ h0f, const float* __restrict__ c0f,
                           const float* __restrict__ h0b, const float* __restrict__ c0b,
                           unsigned short* __restrict__ hf, unsigned short* __restrict__ hb,
                           float* __restrict__ cf, float* __restrict__ cb) {
    int i = blockIdx.x * blockDim.x + threadIdx.x;   // 16384 = B*H
    if (i >= B_ * H_) return;
    int j = i & (H_ - 1);
    hf[i] = f32_to_bf16(h0f[j]);
    hb[i] = f32_to_bf16(h0b[j]);
    cf[i] = c0f[j];
    cb[i] = c0b[j];
}

// ---------------------------------------------------------------- x @ Wi^T + bi
// A = x_bf16 rows (bwd: sent_len-reversed gather), [M=16384, K=1024]
// W = Wi bf16 [N=1536, K=1024] (row-major, K contiguous) -> out [M,1536] f32.
// Block tile 32M x 128N, 8 waves (2x4), two 16x16 accums per wave.
// K staged 64-wide: B tiles via TDM (double-buffered, padded rows),
// A tiles via b128 loads (double-buffered, same padded stride).
__global__ __launch_bounds__(256)
void xproj_gemm(const unsigned short* __restrict__ xbf,
                const unsigned short* __restrict__ Wif, const unsigned short* __restrict__ Wib,
                const float* __restrict__ bif, const float* __restrict__ bib,
                const int* __restrict__ sent_len,
                float* __restrict__ outf, float* __restrict__ outb) {
    const int dir = blockIdx.z;
    const unsigned short* W = dir ? Wib : Wif;
    const float* bi         = dir ? bib : bif;
    float* outp             = dir ? outb : outf;

    const int r0 = blockIdx.x * 32;    // gridDim.x = 512
    const int n0 = blockIdx.y * 128;   // gridDim.y = 12

    __shared__ unsigned As[2][32 * LROW];    // 32 rows x 64 K (+pad), ping-pong
    __shared__ unsigned Bs[2][128 * LROW];   // 128 rows x 64 K (+pad), TDM ping-pong
    __shared__ unsigned Aoff[32];            // gathered row byte offsets into xbf

    const int tid = threadIdx.x;
    if (tid < 32) {
        int r = r0 + tid;
        int b = r >> 9, t = r & (S_ - 1);
        int tt = t;
        if (dir) { int L = sent_len[b]; tt = (t < L) ? (L - 1 - t) : t; }
        Aoff[tid] = (unsigned)((b * S_ + tt) * (I_ * 2));   // bytes
    }
    __syncthreads();

    const int lane = tid & 31;
    const int w    = tid >> 5;   // 0..7
    const int mw   = w >> 2;     // M sub-tile
    const int nw   = w & 3;      // N sub-tile group (32 wide)

    const unsigned ldsB0 = (unsigned)(size_t)(void*)&Bs[0][0];
    const unsigned ldsB1 = (unsigned)(size_t)(void*)&Bs[1][0];
    const u64 gB = (u64)(size_t)W + (u64)n0 * (I_ * 2);     // tile row n0, k=0 (bytes)

    const int NCH = I_ / 64;     // 16 K-chunks of 64

    if (tid < 32)  // prologue: chunk 0 -> buf 0
        tdm_load_2d_bf16(ldsB0, gB, 64, 128, I_, G3, I_);

    v8f acc0 = {}; v8f acc1 = {};

    for (int i = 0; i < NCH; ++i) {
        const int kc = i * 64;
        // A chunk i: 32 rows x 64 bf16 = 256 b128 loads, one per thread
        const int arow = tid >> 3, aq = tid & 7;
        uint4 areg = *(const uint4*)((const char*)xbf + Aoff[arow] + kc * 2 + aq * 16);
        if (tid < 32) {
            if (i + 1 < NCH) {
                tdm_load_2d_bf16((i & 1) ? ldsB0 : ldsB1, gB + (u64)(kc + 64) * 2,
                                 64, 128, I_, G3, I_);
                __builtin_amdgcn_s_wait_tensorcnt(1);   // chunk i resident, i+1 in flight
            } else {
                __builtin_amdgcn_s_wait_tensorcnt(0);
            }
        }
        ((uint4*)As[i & 1])[arow * (LROW / 4) + aq] = areg;
        __syncthreads();

        const unsigned* Asb = As[i & 1] + (mw * 16 + (lane & 15)) * LROW;
        const unsigned* Bsb0 = Bs[i & 1] + (nw * 32 + (lane & 15)) * LROW;
        const unsigned* Bsb1 = Bsb0 + 16 * LROW;
#pragma unroll
        for (int s = 0; s < 2; ++s) {           // two 32-deep WMMA K-steps
            v16bf af = load_frag(Asb,  s * 16);
            v16bf b0 = load_frag(Bsb0, s * 16);
            v16bf b1 = load_frag(Bsb1, s * 16);
            acc0 = wmma_bf16(af, b0, acc0);
            acc1 = wmma_bf16(af, b1, acc1);
        }
        __syncthreads();
    }

    // store + bias. C layout: lanes 0-15: M=v, N=lane; lanes 16-31: M=8+v, N=lane-16
    const int nn    = lane & 15;
    const int mg    = (lane >> 4) * 8;
    const int ncol0 = n0 + nw * 32 + nn;
    const int ncol1 = ncol0 + 16;
    const float bv0 = bi[ncol0];
    const float bv1 = bi[ncol1];
#pragma unroll
    for (int v = 0; v < 8; ++v) {
        int r = r0 + mw * 16 + mg + v;
        outp[(size_t)r * G3 + ncol0] = acc0[v] + bv0;
        outp[(size_t)r * G3 + ncol1] = acc1[v] + bv1;
    }
}

// ---------------------------------------------------------------- recurrent step
// grid = (16, 2): x = 32-wide gate-column chunk, y = direction. 384 thr = 12 waves,
// one 16x16 tile each: mt = batch half, grp = gate (i/j/o), ns = n sub-tile.
// G = xproj[:,t,:] + bh + h_in @ Wh^T ; gates -> c, h. h double-buffered across steps.
__global__ __launch_bounds__(384)
void lstm_step(const float* __restrict__ xprojf, const float* __restrict__ xprojb,
               const unsigned short* __restrict__ Whf, const unsigned short* __restrict__ Whb,
               const float* __restrict__ bhf, const float* __restrict__ bhb,
               const unsigned short* __restrict__ hinf, unsigned short* __restrict__ houtf,
               const unsigned short* __restrict__ hinb, unsigned short* __restrict__ houtb,
               float* __restrict__ cf, float* __restrict__ cb,
               const int* __restrict__ sent_len,
               float* __restrict__ out, int t) {
    const int dir = blockIdx.y;
    const float* xproj   = dir ? xprojb : xprojf;
    const unsigned* Wh   = (const unsigned*)(dir ? Whb : Whf);
    const float* bh      = dir ? bhb : bhf;
    const unsigned* hin  = (const unsigned*)(dir ? hinb : hinf);
    unsigned short* hout = dir ? houtb : houtf;
    float* cst           = dir ? cb : cf;

    const int n0 = blockIdx.x * 32;

    __shared__ float G[32 * 96];   // [b][grp*32 + nn]

    const int tid  = threadIdx.x;
    const int lane = tid & 31;
    const int w    = tid >> 5;     // 0..11
    const int mt   = w & 1;
    const int tn   = w >> 1;       // 0..5
    const int grp  = tn >> 1;      // gate 0..2
    const int ns   = tn & 1;

    const int b0   = mt * 16;
    const int ncol = grp * H_ + n0 + ns * 16 + (lane & 15);
    const int mg   = (lane >> 4) * 8;

    // accumulator init from x-projection + bh
    v8f acc;
    const float bv = bh[ncol];
#pragma unroll
    for (int v = 0; v < 8; ++v) {
        int b = b0 + mg + v;
        acc[v] = xproj[(size_t)(b * S_ + t) * G3 + ncol] + bv;
    }

    // h_in @ Wh^T over K = 512 (fragments via b128 global loads; all L2-resident)
    const unsigned* Arow = hin + (b0 + (lane & 15)) * (H_ / 2);
    const unsigned* Brow = Wh + (size_t)ncol * (H_ / 2);
#pragma unroll 4
    for (int k = 0; k < H_; k += 32) {
        v16bf a  = load_frag(Arow, k >> 1);
        v16bf bm = load_frag(Brow, k >> 1);
        acc = wmma_bf16(a, bm, acc);
    }

    // spill tile to LDS gate buffer
    {
        int col = grp * 32 + ns * 16 + (lane & 15);
#pragma unroll
        for (int v = 0; v < 8; ++v)
            G[(b0 + mg + v) * 96 + col] = acc[v];
    }
    __syncthreads();

    // gate math: 32 batches x 32 columns (<= 3 bounded iterations)
#pragma unroll
    for (int it = 0; it < 3; ++it) {
        int idx = tid + it * 384;
        if (idx < 32 * 32) {
            int b  = idx >> 5;
            int nn = idx & 31;
            float ig = sigmoidf_(G[b * 96 + nn]);
            float jt = tanhf_(G[b * 96 + 32 + nn]);
            float os = sigmoidf_(G[b * 96 + 64 + nn]);
            int n = n0 + nn;
            float c = cst[b * H_ + n];
            c = (1.0f - ig) * c + ig * jt;
            float h = tanhf_(c) * os;
            cst[b * H_ + n] = c;
            hout[b * H_ + n] = f32_to_bf16(h);
            int to = t;
            if (dir) { int L = sent_len[b]; to = (t < L) ? (L - 1 - t) : t; }
            out[(size_t)(b * S_ + to) * (2 * H_) + dir * H_ + n] = h;
        }
    }
}

// ---------------------------------------------------------------- launch
extern "C" void kernel_launch(void* const* d_in, const int* in_sizes, int n_in,
                              void* d_out, int out_size, void* d_ws, size_t ws_size,
                              hipStream_t stream) {
    const float* x        = (const float*)d_in[0];
    const int*   sent_len = (const int*)d_in[1];
    const float* fwd_h0   = (const float*)d_in[2];
    const float* fwd_c0   = (const float*)d_in[3];
    const float* fwd_Wi   = (const float*)d_in[4];
    const float* fwd_bi   = (const float*)d_in[5];
    const float* fwd_Wh   = (const float*)d_in[6];
    const float* fwd_bh   = (const float*)d_in[7];
    const float* bwd_h0   = (const float*)d_in[8];
    const float* bwd_c0   = (const float*)d_in[9];
    const float* bwd_Wi   = (const float*)d_in[10];
    const float* bwd_bi   = (const float*)d_in[11];
    const float* bwd_Wh   = (const float*)d_in[12];
    const float* bwd_bh   = (const float*)d_in[13];
    float* out = (float*)d_out;

    char* p = (char*)d_ws;
    auto take = [&](size_t bytes) {
        char* q = p;
        p += (bytes + 255) & ~(size_t)255;
        return q;
    };
    float*          xproj_f = (float*)take((size_t)MROWS * G3 * 4);
    float*          xproj_b = (float*)take((size_t)MROWS * G3 * 4);
    unsigned short* x_bf    = (unsigned short*)take((size_t)MROWS * I_ * 2);
    unsigned short* Wif_bf  = (unsigned short*)take((size_t)G3 * I_ * 2);
    unsigned short* Wib_bf  = (unsigned short*)take((size_t)G3 * I_ * 2);
    unsigned short* Whf_bf  = (unsigned short*)take((size_t)G3 * H_ * 2);
    unsigned short* Whb_bf  = (unsigned short*)take((size_t)G3 * H_ * 2);
    unsigned short* hf0     = (unsigned short*)take((size_t)B_ * H_ * 2);
    unsigned short* hf1     = (unsigned short*)take((size_t)B_ * H_ * 2);
    unsigned short* hb0     = (unsigned short*)take((size_t)B_ * H_ * 2);
    unsigned short* hb1     = (unsigned short*)take((size_t)B_ * H_ * 2);
    float*          cfst    = (float*)take((size_t)B_ * H_ * 4);
    float*          cbst    = (float*)take((size_t)B_ * H_ * 4);

    cvt_f32_bf16<<<4096, 256, 0, stream>>>(x,      x_bf,   MROWS * I_);
    cvt_f32_bf16<<<1024, 256, 0, stream>>>(fwd_Wi, Wif_bf, G3 * I_);
    cvt_f32_bf16<<<1024, 256, 0, stream>>>(bwd_Wi, Wib_bf, G3 * I_);
    cvt_f32_bf16<<<512,  256, 0, stream>>>(fwd_Wh, Whf_bf, G3 * H_);
    cvt_f32_bf16<<<512,  256, 0, stream>>>(bwd_Wh, Whb_bf, G3 * H_);
    init_state<<<(B_ * H_ + 255) / 256, 256, 0, stream>>>(fwd_h0, fwd_c0, bwd_h0, bwd_c0,
                                                          hf0, hb0, cfst, cbst);

    xproj_gemm<<<dim3(MROWS / 32, G3 / 128, 2), 256, 0, stream>>>(
        x_bf, Wif_bf, Wib_bf, fwd_bi, bwd_bi, sent_len, xproj_f, xproj_b);

    for (int t = 0; t < S_; ++t) {
        const unsigned short* hin_f  = (t & 1) ? hf1 : hf0;
        unsigned short*       hout_f = (t & 1) ? hf0 : hf1;
        const unsigned short* hin_b  = (t & 1) ? hb1 : hb0;
        unsigned short*       hout_b = (t & 1) ? hb0 : hb1;
        lstm_step<<<dim3(H_ / 32, 2), 384, 0, stream>>>(
            xproj_f, xproj_b, Whf_bf, Whb_bf, fwd_bh, bwd_bh,
            hin_f, hout_f, hin_b, hout_b, cfst, cbst, sent_len, out, t);
    }
}
